// QuantumDiscriminator_71476845740522
// MI455X (gfx1250) — compile-verified, matrix operand order checked by reading, not run on hardware
//
#include <hip/hip_runtime.h>
#include <hip/hip_bf16.h>

typedef __bf16 bf16;
typedef __attribute__((ext_vector_type(16))) __bf16 v16bf;
typedef __attribute__((ext_vector_type(8)))  __bf16 v8bf;
typedef __attribute__((ext_vector_type(4)))  __bf16 v4bf;
typedef __attribute__((ext_vector_type(8)))  float  v8f;
typedef __attribute__((ext_vector_type(4)))  int    v4i;

// BK=64: row = 64 bf16 data + pad -> stride 72 elems (144 B, odd multiple of 16B)
#define ASTRIDE 72
#define ABUF (128*ASTRIDE)   // elements per A buffer
#define BBUF ( 64*ASTRIDE)   // elements per B buffer

#if defined(__AMDGCN__) && __has_builtin(__builtin_amdgcn_global_load_async_to_lds_b128)
#define HAVE_ASYNC_LDS 1
typedef __attribute__((address_space(1))) v4i* gptr_v4i;
typedef __attribute__((address_space(3))) v4i* lptr_v4i;
__device__ __forceinline__ void async_copy16(const bf16* g, bf16* l) {
  // GLOBAL_LOAD_ASYNC_TO_LDS_B128: 16 bytes, tracked by ASYNCcnt
  __builtin_amdgcn_global_load_async_to_lds_b128((gptr_v4i)g, (lptr_v4i)l, 0, 0);
}
#else
#define HAVE_ASYNC_LDS 0
#endif

__device__ __forceinline__ float leaky_f(float x){ return x >= 0.f ? x : 0.2f*x; }

// ---------------------------------------------------------------- f32 -> bf16
__global__ void cvt_bf16_kernel(const float* __restrict__ src, bf16* __restrict__ dst, int n){
  int i = blockIdx.x*blockDim.x + threadIdx.x;
  int stride = gridDim.x*blockDim.x;
  for (; i < n; i += stride) dst[i] = (bf16)src[i];
}

// ------------------------------------------------------- WMMA bf16 GEMM + leaky
// C[M,N] = leaky(A[M,K] (f32) @ Bw[N,K]^T (bf16) + bias[N])
// Block tile 128x64, K-step 64, 256 threads (8 waves), wave tile 32x32.
// Double-buffered LDS (index-offset addressing keeps AS3 GEPs -> ds ops);
// B tile staged with async global->LDS.
__global__ __launch_bounds__(256) void gemm_bf16_leaky_kernel(
    const float* __restrict__ A, const bf16* __restrict__ Bw,
    const float* __restrict__ bias, float* __restrict__ C,
    int M, int N, int K)
{
  __shared__ bf16 As[2*ABUF];   // 2 x 18 KB
  __shared__ bf16 Bs[2*BBUF];   // 2 x  9 KB
  const int tid  = threadIdx.x;
  const int lane = tid & 31;
  const int wave = tid >> 5;
  const int wm = wave >> 1;      // 0..3  (32-row slab)
  const int wn = wave & 1;       // 0..1  (32-col slab)
  const int row0 = blockIdx.x * 128;
  const int col0 = blockIdx.y * 64;

  // staging assignment
  const int arow = tid & 127;          // A: row 0..127
  const int ahalf = tid >> 7;          // A: k-half 0/1 (32 floats each)
  const int brow = tid >> 2;           // B: row 0..63
  const int bseg = tid & 3;            // B: 32B segment 0..3 (16 bf16)

  const float* Ap = A + (size_t)(row0 + arow)*K + ahalf*32;
  const bf16*  Bp = Bw + (size_t)(col0 + brow)*K + bseg*16;
  const int aOff = arow*ASTRIDE + ahalf*32;   // element offset within an A buffer
  const int bOff = brow*ASTRIDE + bseg*16;    // element offset within a B buffer

  v8f acc[2][2] = {};

  // ---- prologue: stage tile 0 into buffer 0
  {
    float4 f[8];
    #pragma unroll
    for (int i = 0; i < 8; ++i) f[i] = *(const float4*)(Ap + i*4);       // pipelined
#if HAVE_ASYNC_LDS
    async_copy16(Bp,     &Bs[bOff]);
    async_copy16(Bp + 8, &Bs[bOff + 8]);
#else
    v8bf b0 = *(const v8bf*)Bp, b1 = *(const v8bf*)(Bp + 8);
    *(v8bf*)&Bs[bOff] = b0;  *(v8bf*)&Bs[bOff + 8] = b1;
#endif
    #pragma unroll
    for (int c = 0; c < 4; ++c) {
      float4 x = f[2*c], y = f[2*c + 1];
      v8bf h = { (bf16)x.x,(bf16)x.y,(bf16)x.z,(bf16)x.w,
                 (bf16)y.x,(bf16)y.y,(bf16)y.z,(bf16)y.w };
      *(v8bf*)&As[aOff + c*8] = h;
    }
#if HAVE_ASYNC_LDS
    asm volatile("s_wait_asynccnt 0x0" ::: "memory");
#endif
  }
  __syncthreads();

  const int nk = K >> 6;
  const int klo = (lane >> 4) * 8;     // ISA 16-bit A/B layout: lanes>=16 take K lo 8

  for (int it = 0; it < nk; ++it) {
    const int cur = it & 1;
    const int nxt = cur ^ 1;
    const bool more = (it + 1) < nk;
    const int aCur = cur*ABUF, bCur = cur*BBUF;
    const int aNxt = nxt*ABUF, bNxt = nxt*BBUF;

    // ---- issue next-tile global loads early (overlap with WMMA below)
    float4 f[8];
    if (more) {
      const float* ap = Ap + (it + 1)*64;
      #pragma unroll
      for (int i = 0; i < 8; ++i) f[i] = *(const float4*)(ap + i*4);
#if HAVE_ASYNC_LDS
      const bf16* bp = Bp + (it + 1)*64;
      async_copy16(bp,     &Bs[bNxt + bOff]);
      async_copy16(bp + 8, &Bs[bNxt + bOff + 8]);
#endif
      if ((it + 2) < nk)
        __builtin_prefetch(Ap + (it + 2)*64, 0, 1);
    }

    // ---- compute on buffer `cur` (2 k-blocks of 32)
    #pragma unroll
    for (int kb = 0; kb < 2; ++kb) {
      const int kbase = kb*32 + klo;
      #pragma unroll
      for (int i = 0; i < 2; ++i) {
        union { v16bf v; v8bf h[2]; } af;
        const int ar = wm*32 + i*16 + (lane & 15);
        af.h[0] = *(const v8bf*)&As[aCur + ar*ASTRIDE + kbase];
        af.h[1] = *(const v8bf*)&As[aCur + ar*ASTRIDE + kbase + 16];
        #pragma unroll
        for (int j = 0; j < 2; ++j) {
          union { v16bf v; v8bf h[2]; } bfg;
          const int br = wn*32 + j*16 + (lane & 15);
          bfg.h[0] = *(const v8bf*)&Bs[bCur + br*ASTRIDE + kbase];
          bfg.h[1] = *(const v8bf*)&Bs[bCur + br*ASTRIDE + kbase + 16];
          acc[i][j] = __builtin_amdgcn_wmma_f32_16x16x32_bf16(
              false, af.v, false, bfg.v, (short)0, acc[i][j], false, false);
        }
      }
    }

    // ---- commit next tile into buffer `nxt`
    if (more) {
      #pragma unroll
      for (int c = 0; c < 4; ++c) {
        float4 x = f[2*c], y = f[2*c + 1];
        v8bf h = { (bf16)x.x,(bf16)x.y,(bf16)x.z,(bf16)x.w,
                   (bf16)y.x,(bf16)y.y,(bf16)y.z,(bf16)y.w };
        *(v8bf*)&As[aNxt + aOff + c*8] = h;
      }
#if !HAVE_ASYNC_LDS
      const bf16* bp = Bp + (it + 1)*64;
      v8bf b0 = *(const v8bf*)bp, b1 = *(const v8bf*)(bp + 8);
      *(v8bf*)&Bs[bNxt + bOff] = b0;  *(v8bf*)&Bs[bNxt + bOff + 8] = b1;
#else
      asm volatile("s_wait_asynccnt 0x0" ::: "memory");
#endif
    }
    __syncthreads();
  }

  // epilogue: C/D layout — VGPR v holds M = v (lanes 0-15) / 8+v (lanes 16-31)
  #pragma unroll
  for (int i = 0; i < 2; ++i)
    #pragma unroll
    for (int j = 0; j < 2; ++j) {
      const int col = col0 + wn*32 + j*16 + (lane & 15);
      const float bb = bias[col];
      #pragma unroll
      for (int v = 0; v < 8; ++v) {
        const int row = row0 + wm*32 + i*16 + v + 8*(lane >> 4);
        C[(size_t)row*N + col] = leaky_f(acc[i][j][v] + bb);
      }
    }
}

// -------------------------------------------------------------- quantum sim
// state index a[7:0]; qubit q lives in bit p = 7-q. lane = a[7:3], reg = a[2:0].
__device__ __forceinline__ float2 cmul(float2 a, float2 b){
  return make_float2(a.x*b.x - a.y*b.y, a.x*b.y + a.y*b.x);
}
__device__ __forceinline__ float2 cadd(float2 a, float2 b){
  return make_float2(a.x + b.x, a.y + b.y);
}

template<int P>
__device__ __forceinline__ void apply1q(float2 (&amp)[8], float2 u00, float2 u01,
                                        float2 u10, float2 u11, int lane)
{
  if constexpr (P < 3) {
    #pragma unroll
    for (int r = 0; r < 8; ++r) {
      if ((r & (1 << P)) == 0) {
        const int r1 = r | (1 << P);
        float2 a0 = amp[r], a1 = amp[r1];
        amp[r]  = cadd(cmul(u00, a0), cmul(u01, a1));
        amp[r1] = cadd(cmul(u10, a0), cmul(u11, a1));
      }
    }
  } else {
    const int m = 1 << (P - 3);
    const bool hi = (lane & m) != 0;
    #pragma unroll
    for (int r = 0; r < 8; ++r) {
      float2 o;
      o.x = __shfl_xor(amp[r].x, m, 32);
      o.y = __shfl_xor(amp[r].y, m, 32);
      float2 a = amp[r];
      float2 lov = cadd(cmul(u00, a), cmul(u01, o));
      float2 hiv = cadd(cmul(u10, o), cmul(u11, a));
      amp[r] = hi ? hiv : lov;
    }
  }
}

template<int P>
__device__ __forceinline__ void applyRZ(float2 (&amp)[8], float c, float s, int lane)
{
  if constexpr (P < 3) {
    #pragma unroll
    for (int r = 0; r < 8; ++r) {
      float sg = ((r >> P) & 1) ? s : -s;
      amp[r] = cmul(amp[r], make_float2(c, sg));
    }
  } else {
    float sg = ((lane >> (P - 3)) & 1) ? s : -s;
    #pragma unroll
    for (int r = 0; r < 8; ++r) amp[r] = cmul(amp[r], make_float2(c, sg));
  }
}

template<int PC, int PT>
__device__ __forceinline__ void cnot_g(float2 (&amp)[8], int lane)
{
  if constexpr (PT >= 3) {            // control & target in lane bits
    const int tm = 1 << (PT - 3);
    const bool ctrl = (lane >> (PC - 3)) & 1;
    #pragma unroll
    for (int r = 0; r < 8; ++r) {
      float2 o;
      o.x = __shfl_xor(amp[r].x, tm, 32);
      o.y = __shfl_xor(amp[r].y, tm, 32);
      if (ctrl) amp[r] = o;
    }
  } else if constexpr (PC >= 3) {     // PC=3 (lane bit 0), PT=2 (reg bit 2)
    const bool ctrl = (lane >> (PC - 3)) & 1;
    #pragma unroll
    for (int r = 0; r < 4; ++r) {
      if (ctrl) { float2 t = amp[r]; amp[r] = amp[r | 4]; amp[r | 4] = t; }
    }
  } else {                            // both in reg bits
    #pragma unroll
    for (int r = 0; r < 8; ++r) {
      if ((r & (1 << PC)) && !(r & (1 << PT))) {
        const int r1 = r | (1 << PT);
        float2 t = amp[r]; amp[r] = amp[r1]; amp[r1] = t;
      }
    }
  }
}

#define Q_INIT(q) { float t = ang[q]*0.5f; float c = cosf(t), s = sinf(t); \
  apply1q<7-(q)>(amp, make_float2(c,0.f), make_float2(0.f,-s), \
                 make_float2(0.f,-s), make_float2(c,0.f), lane); }

#define Q_LAYER(q) { \
  float tx = qp[(l*8+(q))*3+0]*0.5f; float cx = cosf(tx), sx = sinf(tx); \
  apply1q<7-(q)>(amp, make_float2(cx,0.f), make_float2(0.f,-sx), \
                 make_float2(0.f,-sx), make_float2(cx,0.f), lane); \
  float ty = qp[(l*8+(q))*3+1]*0.5f; float cy = cosf(ty), sy = sinf(ty); \
  apply1q<7-(q)>(amp, make_float2(cy,0.f), make_float2(-sy,0.f), \
                 make_float2(sy,0.f), make_float2(cy,0.f), lane); \
  float tz = qp[(l*8+(q))*3+2]*0.5f; float cz = cosf(tz), sz = sinf(tz); \
  applyRZ<7-(q)>(amp, cz, sz, lane); }

#define FOR_ALL_Q(OP) OP(0) OP(1) OP(2) OP(3) OP(4) OP(5) OP(6) OP(7)

__global__ __launch_bounds__(256) void quantum_kernel(
    const float* __restrict__ h2, const float* __restrict__ W3,
    const float* __restrict__ b3, const float* __restrict__ qp,
    float* __restrict__ qf, int B)
{
  const int lane = threadIdx.x & 31;
  const int wave = threadIdx.x >> 5;
  const int b = blockIdx.x*8 + wave;
  if (b >= B) return;

  float x[8];
  #pragma unroll
  for (int j = 0; j < 8; ++j) x[j] = h2[(size_t)b*256 + lane + 32*j];
  float ang[8];
  #pragma unroll
  for (int n = 0; n < 8; ++n) {
    float p = 0.f;
    #pragma unroll
    for (int j = 0; j < 8; ++j) p += x[j]*W3[n*256 + lane + 32*j];
    #pragma unroll
    for (int off = 16; off >= 1; off >>= 1) p += __shfl_xor(p, off, 32);
    ang[n] = (p + b3[n] + 1.0f) * 1.5707963267948966f;
  }

  float2 amp[8];
  #pragma unroll
  for (int r = 0; r < 8; ++r) amp[r] = make_float2(0.f, 0.f);
  if (lane == 0) amp[0].x = 1.0f;

  FOR_ALL_Q(Q_INIT)

  #pragma unroll
  for (int l = 0; l < 2; ++l) {
    FOR_ALL_Q(Q_LAYER)
    cnot_g<7,6>(amp, lane); cnot_g<6,5>(amp, lane);
    cnot_g<5,4>(amp, lane); cnot_g<4,3>(amp, lane);
    cnot_g<3,2>(amp, lane); cnot_g<2,1>(amp, lane);
    cnot_g<1,0>(amp, lane);
  }

  float pr[8]; float tot = 0.f;
  #pragma unroll
  for (int r = 0; r < 8; ++r) { pr[r] = amp[r].x*amp[r].x + amp[r].y*amp[r].y; tot += pr[r]; }
  #pragma unroll
  for (int q = 0; q < 8; ++q) {
    const int p = 7 - q;
    float loc;
    if (p < 3) {
      loc = 0.f;
      #pragma unroll
      for (int r = 0; r < 8; ++r) loc += ((r >> p) & 1) ? -pr[r] : pr[r];
    } else {
      loc = ((lane >> (p - 3)) & 1) ? -tot : tot;
    }
    #pragma unroll
    for (int off = 16; off >= 1; off >>= 1) loc += __shfl_xor(loc, off, 32);
    if (lane == 0) qf[(size_t)b*8 + q] = loc;
  }
}

// ------------------------------------------------------------------ MLP head
__global__ __launch_bounds__(256) void head_kernel(
    const float* __restrict__ qf, const float* __restrict__ emb,
    const int* __restrict__ cls,
    const float* __restrict__ W4, const float* __restrict__ b4,
    const float* __restrict__ W5, const float* __restrict__ b5,
    const float* __restrict__ W6, const float* __restrict__ b6,
    float* __restrict__ out)
{
  __shared__ float h4s[8][256];
  const int lane = threadIdx.x & 31;
  const int wave = threadIdx.x >> 5;
  const int b = blockIdx.x*8 + wave;

  float comb[14];
  #pragma unroll
  for (int k = 0; k < 8; ++k) comb[k] = qf[(size_t)b*8 + k];
  const int cid = cls[b];
  #pragma unroll
  for (int k = 0; k < 6; ++k) comb[8 + k] = emb[cid*6 + k];

  #pragma unroll
  for (int u = 0; u < 8; ++u) {
    const int n = lane*8 + u;
    float a = b4[n];
    #pragma unroll
    for (int k = 0; k < 14; ++k) a += comb[k]*W4[n*14 + k];
    h4s[wave][n] = leaky_f(a);
  }
  __syncthreads();

  float part = 0.f;
  #pragma unroll
  for (int v = 0; v < 4; ++v) {
    const int m = lane*4 + v;
    float a = b5[m];
    for (int n = 0; n < 256; ++n) a += h4s[wave][n]*W5[m*256 + n];
    part += leaky_f(a)*W6[m];
  }
  #pragma unroll
  for (int off = 16; off >= 1; off >>= 1) part += __shfl_xor(part, off, 32);
  if (lane == 0) out[b] = part + b6[0];
}

// ------------------------------------------------------------------- launch
extern "C" void kernel_launch(void* const* d_in, const int* in_sizes, int n_in,
                              void* d_out, int out_size, void* d_ws, size_t ws_size,
                              hipStream_t stream) {
  (void)in_sizes; (void)n_in; (void)out_size; (void)ws_size;
  const float* img = (const float*)d_in[0];
  const int*   cls = (const int*)  d_in[1];
  const float* W1  = (const float*)d_in[2];
  const float* b1  = (const float*)d_in[3];
  const float* W2  = (const float*)d_in[4];
  const float* b2  = (const float*)d_in[5];
  const float* W3  = (const float*)d_in[6];
  const float* b3  = (const float*)d_in[7];
  const float* qp  = (const float*)d_in[8];
  const float* emb = (const float*)d_in[9];
  const float* W4  = (const float*)d_in[10];
  const float* b4  = (const float*)d_in[11];
  const float* W5  = (const float*)d_in[12];
  const float* b5  = (const float*)d_in[13];
  const float* W6  = (const float*)d_in[14];
  const float* b6  = (const float*)d_in[15];
  float* out = (float*)d_out;

  char* ws = (char*)d_ws;
  bf16*  w1bf = (bf16*)(ws);                // 256*16384*2 = 8 MB
  bf16*  w2bf = (bf16*)(ws + 8388608);      // 256*256*2   = 128 KB
  float* h1   = (float*)(ws + 8519680);     // 4096*256*4  = 4 MB
  float* h2   = (float*)(ws + 12713984);    // 4 MB
  float* qfb  = (float*)(ws + 16908288);    // 4096*8*4    = 128 KB

  cvt_bf16_kernel<<<2048, 256, 0, stream>>>(W1, w1bf, 256*16384);
  cvt_bf16_kernel<<<64,   256, 0, stream>>>(W2, w2bf, 256*256);

  dim3 g1(4096/128, 256/64);
  gemm_bf16_leaky_kernel<<<g1, 256, 0, stream>>>(img, w1bf, b1, h1, 4096, 256, 16384);
  gemm_bf16_leaky_kernel<<<g1, 256, 0, stream>>>(h1,  w2bf, b2, h2, 4096, 256, 256);

  quantum_kernel<<<512, 256, 0, stream>>>(h2, W3, b3, qp, qfb, 4096);
  head_kernel<<<512, 256, 0, stream>>>(qfb, emb, cls, W4, b4, W5, b5, W6, b6, out);
}
